// ObstacleSelector_80573586473319
// MI455X (gfx1250) — matrix operand order repaired and checked by reading, not compiled, not run on hardware
//
#include <hip/hip_runtime.h>
#include <math.h>

// ---------------- problem constants ----------------
#define BB    4
#define NN    10
#define NIMG  44          // 4 target + 40 object images through resnet
#define IMGSZ 224
#define HWPX  (IMGSZ*IMGSZ)
#define HID   1024
#define DIMF  512

typedef unsigned short ushort_t;

// ---------------- WMMA types / helpers ----------------
typedef __attribute__((ext_vector_type(16))) __bf16 v16bf;
typedef __attribute__((ext_vector_type(8)))  float  v8f;

__device__ __forceinline__ ushort_t f2bf_bits(float f) {
    // round-to-nearest-even f32 -> bf16 bits (finite inputs)
    unsigned u = __builtin_bit_cast(unsigned, f);
    u += 0x7FFFu + ((u >> 16) & 1u);
    return (ushort_t)(u >> 16);
}
__device__ __forceinline__ __bf16 f2bf(float f) {
    return __builtin_bit_cast(__bf16, f2bf_bits(f));
}
__device__ __forceinline__ float bf2f(ushort_t h) {
    unsigned u = (unsigned)h << 16;
    return __builtin_bit_cast(float, u);
}
__device__ __forceinline__ __bf16 bfbits(ushort_t h) {
    return __builtin_bit_cast(__bf16, h);
}

struct bfrag32 { uint4 lo, hi; };   // 32B: elements 0..7, 8..15 of a v16bf
union pk8 { ushort_t u[8]; uint4 v; };

__device__ __forceinline__ v8f wmma_bf16(const v16bf& a, const v16bf& b, const v8f& c) {
    return __builtin_amdgcn_wmma_f32_16x16x32_bf16(false, a, false, b, (short)0, c, false, false);
}

// ---------------- weight reorder + convert: OIHW f32 -> (O,kh,kw,I) bf16 ----------------
__global__ void wreorder_kernel(const float* __restrict__ src, ushort_t* __restrict__ dst,
                                int Cout, int Cin, int KS)
{
    long n = (long)Cout * Cin * KS * KS;
    long i = (long)blockIdx.x * blockDim.x + threadIdx.x;
    if (i >= n) return;
    int ci = (int)(i % Cin); long t = i / Cin;
    int kw = (int)(t % KS); t /= KS;
    int kh = (int)(t % KS); int co = (int)(t / KS);
    dst[i] = f2bf_bits(src[((long)(co*Cin + ci)*KS + kh)*KS + kw]);
}

// =====================================================================
// Implicit-GEMM conv, NHWC bf16 activations, (O,kh,kw,I) bf16 weights,
// fused BN (+residual)(+ReLU), f32 accumulate.
// One wave: 16(cout) x 64(pixel) tile, 4 v_wmma per 32-wide K-step.
// K order = (kh, kw, cin), cin fastest. For Cin%16==0 a 16-k B run is
// 16 contiguous bf16 -> two b128 loads + one bounds check per tile.
// =====================================================================
template<int KS>
__global__ void conv_bn_wmma(const ushort_t* __restrict__ in,
                             const ushort_t* __restrict__ wgt,
                             const float* __restrict__ bg,
                             const float* __restrict__ bb,
                             const float* __restrict__ bm,
                             const float* __restrict__ bv,
                             const ushort_t* __restrict__ res,
                             ushort_t* __restrict__ out,
                             int Cin, int cinShift, int H, int W,
                             int Cout, int Ho, int Wo,
                             int stride, int pad, int relu)
{
    const int lane  = threadIdx.x & 31;
    const int n0    = blockIdx.x * 64;          // pixel tile base (4 x 16)
    const int m0    = blockIdx.y * 16;          // cout tile base
    const int Npix  = NIMG * Ho * Wo;
    const int K     = Cin * KS * KS;
    const int Kmain = K & ~31;

    v8f acc[4];
    #pragma unroll
    for (int j = 0; j < 4; ++j) acc[j] = (v8f){0.f,0.f,0.f,0.f,0.f,0.f,0.f,0.f};

    // ---- A side: clamped row (garbage rows never stored) ----
    const int mAr = m0 + (lane & 15);
    const int mA  = (mAr < Cout) ? mAr : (Cout - 1);
    const ushort_t* wrow = wgt + (long)mA * K;
    const int kbaseA  = (lane >> 4) * 16;

    // ---- B side: 4 pixel columns, clamped decode ----
    const int kbaseB = (lane >> 4) * 16;
    int img[4], oh[4], ow[4], ihb[4], iwb[4];
    bool nval[4];
    #pragma unroll
    for (int j = 0; j < 4; ++j) {
        int nB  = n0 + j*16 + (lane & 15);
        nval[j] = (nB < Npix);
        int ns  = nval[j] ? nB : (Npix - 1);
        int im  = ns / (Ho*Wo); int r = ns % (Ho*Wo);
        img[j] = im; oh[j] = r / Wo; ow[j] = r % Wo;
        ihb[j] = oh[j]*stride - pad; iwb[j] = ow[j]*stride - pad;
    }

    // -------- main K loop --------
    for (int k0 = 0; k0 < Kmain; k0 += 32) {
        if (k0 + 64 < K) __builtin_prefetch(&wrow[k0 + 64], 0, 1);

        v16bf a;
        if constexpr (KS != 7) {
            // K%32==0, rows 16B-aligned -> two raw b128 loads
            bfrag32 fr;
            fr.lo = *(const uint4*)(wrow + k0 + kbaseA);        // K run {0..7}
            fr.hi = *(const uint4*)(wrow + k0 + kbaseA + 8);    // K run {8..15}
            a = __builtin_bit_cast(v16bf, fr);
        } else {
            #pragma unroll
            for (int i = 0; i < 16; ++i) {
                int k = k0 + kbaseA + i;
                a[i] = bfbits(wrow[k]);
            }
        }

        v16bf bf[4];
        if constexpr (KS != 7) {
            // 16-k run stays inside one (kh,kw) cell (Cin%16==0)
            const int kstart = k0 + kbaseB;
            const int cell   = kstart >> cinShift;
            const int cin0   = kstart & (Cin - 1);
            const int kh = cell / KS, kw = cell % KS;
            const uint4 z = {0,0,0,0};
            #pragma unroll
            for (int j = 0; j < 4; ++j) {
                int  ih = ihb[j] + kh, iw = iwb[j] + kw;
                bool ok = (ih >= 0) & (ih < H) & (iw >= 0) & (iw < W);
                long addr = ok ? ((((long)img[j]*H + ih)*W + iw)*Cin + cin0) : 0;
                const uint4* p = (const uint4*)(in + addr);
                uint4 lo = p[0], hi = p[1];
                bfrag32 fr; fr.lo = ok ? lo : z; fr.hi = ok ? hi : z;
                bf[j] = __builtin_bit_cast(v16bf, fr);
            }
        } else {
            // stem: Cin=3, per-element gather
            #pragma unroll
            for (int i = 0; i < 16; ++i) {
                int k   = k0 + kbaseB + i;
                int cell = k / 3; int cin = k - cell*3;
                int kh = cell / 7, kw = cell - (cell/7)*7;
                #pragma unroll
                for (int j = 0; j < 4; ++j) {
                    int  ih = ihb[j] + kh, iw = iwb[j] + kw;
                    bool ok = (ih >= 0) & (ih < H) & (iw >= 0) & (iw < W);
                    long idx = ok ? ((((long)img[j]*H + ih)*W + iw)*3 + cin) : 0;
                    ushort_t v = in[idx];
                    bf[j][i] = bfbits(ok ? v : (ushort_t)0);
                }
            }
        }
        #pragma unroll
        for (int j = 0; j < 4; ++j) acc[j] = wmma_bf16(a, bf[j], acc[j]);
    }

    // -------- K tail: only the stem has K%32 != 0 --------
    if constexpr (KS == 7) {
        if (Kmain < K) {
            v16bf a;
            #pragma unroll
            for (int i = 0; i < 16; ++i) {
                int k  = Kmain + kbaseA + i;
                int ks = (k < K) ? k : (K - 1);
                ushort_t v = wrow[ks];
                a[i] = bfbits((k < K) ? v : (ushort_t)0);
            }
            v16bf bf[4];
            #pragma unroll
            for (int i = 0; i < 16; ++i) {
                int k  = Kmain + kbaseB + i;
                int ks = (k < K) ? k : (K - 1);
                int cell = ks / 3; int cin = ks - cell*3;
                int kh = cell / 7, kw = cell - (cell/7)*7;
                #pragma unroll
                for (int j = 0; j < 4; ++j) {
                    int  ih = ihb[j] + kh, iw = iwb[j] + kw;
                    bool ok = (k < K) & (ih >= 0) & (ih < H) & (iw >= 0) & (iw < W);
                    long idx = ok ? ((((long)img[j]*H + ih)*W + iw)*3 + cin) : 0;
                    ushort_t v = in[idx];
                    bf[j][i] = bfbits(ok ? v : (ushort_t)0);
                }
            }
            #pragma unroll
            for (int j = 0; j < 4; ++j) acc[j] = wmma_bf16(a, bf[j], acc[j]);
        }
    }

    // -------- fused BN (+res)(+relu) epilogue, NHWC: 8 couts contiguous --------
    const int mbase = m0 + ((lane >> 4) << 3);
    float sc[8], sh[8];
    #pragma unroll
    for (int r = 0; r < 8; ++r) {          // Cout%16==0 -> always in range
        int m = mbase + r;
        float inv = bg[m] * rsqrtf(bv[m] + 1e-5f);
        sc[r] = inv; sh[r] = bb[m] - bm[m] * inv;
    }
    #pragma unroll
    for (int j = 0; j < 4; ++j) {
        if (!nval[j]) continue;
        long base = (((long)img[j]*Ho + oh[j])*Wo + ow[j])*Cout + mbase;
        pk8 o;
        if (res) {
            pk8 rv; rv.v = *(const uint4*)(res + base);
            #pragma unroll
            for (int r = 0; r < 8; ++r) {
                float y = acc[j][r]*sc[r] + sh[r] + bf2f(rv.u[r]);
                if (relu) y = fmaxf(y, 0.f);
                o.u[r] = f2bf_bits(y);
            }
        } else {
            #pragma unroll
            for (int r = 0; r < 8; ++r) {
                float y = acc[j][r]*sc[r] + sh[r];
                if (relu) y = fmaxf(y, 0.f);
                o.u[r] = f2bf_bits(y);
            }
        }
        *(uint4*)(out + base) = o.v;
    }
}

// =====================================================================
// Dense GEMM (f32 IO): Y[M][N] = X[M][K] * W[N][K]^T + bias[N]
// =====================================================================
__global__ void gemm_xwT_wmma(const float* __restrict__ X,
                              const float* __restrict__ Wt,
                              const float* __restrict__ bias,
                              float* __restrict__ Y,
                              int M, int N, int K)
{
    const int lane = threadIdx.x & 31;
    const int n0 = blockIdx.x * 64;
    const int m0 = blockIdx.y * 16;
    const int Kmain = K & ~31;

    v8f acc[4];
    #pragma unroll
    for (int j = 0; j < 4; ++j) acc[j] = (v8f){0.f,0.f,0.f,0.f,0.f,0.f,0.f,0.f};

    const int mAr = m0 + (lane & 15);
    const int mA  = (mAr < M) ? mAr : (M - 1);
    const float* xrow = X + (long)mA * K;
    const int kbaseA  = (lane >> 4) * 16;

    const int kbaseB = (lane >> 4) * 16;
    int nB[4]; bool nval[4]; const float* wrow[4];
    #pragma unroll
    for (int j = 0; j < 4; ++j) {
        nB[j]   = n0 + j*16 + (lane & 15);
        nval[j] = (nB[j] < N);
        wrow[j] = Wt + (long)(nval[j] ? nB[j] : (N - 1)) * K;
    }

    for (int k0 = 0; k0 < Kmain; k0 += 32) {
        if (k0 + 64 < K) __builtin_prefetch(&wrow[0][k0 + 64], 0, 1);

        v16bf a;
        {   // K%4==0 for all call sites -> aligned b128 loads
            const float4* p0 = (const float4*)(xrow + k0 + kbaseA);
            const float4* p1 = (const float4*)(xrow + k0 + kbaseA + 8);
            float4 x0 = p0[0], x1 = p0[1], x2 = p1[0], x3 = p1[1];
            a[0]=f2bf(x0.x); a[1]=f2bf(x0.y); a[2] =f2bf(x0.z); a[3] =f2bf(x0.w);
            a[4]=f2bf(x1.x); a[5]=f2bf(x1.y); a[6] =f2bf(x1.z); a[7] =f2bf(x1.w);
            a[8]=f2bf(x2.x); a[9]=f2bf(x2.y); a[10]=f2bf(x2.z); a[11]=f2bf(x2.w);
            a[12]=f2bf(x3.x);a[13]=f2bf(x3.y);a[14] =f2bf(x3.z);a[15] =f2bf(x3.w);
        }
        v16bf bf[4];
        #pragma unroll
        for (int j = 0; j < 4; ++j) {
            const float4* q = (const float4*)(wrow[j] + k0 + kbaseB);
            float4 y0 = q[0], y1 = q[1], y2 = q[2], y3 = q[3];
            bf[j][0] =f2bf(y0.x); bf[j][1] =f2bf(y0.y); bf[j][2] =f2bf(y0.z); bf[j][3] =f2bf(y0.w);
            bf[j][4] =f2bf(y1.x); bf[j][5] =f2bf(y1.y); bf[j][6] =f2bf(y1.z); bf[j][7] =f2bf(y1.w);
            bf[j][8] =f2bf(y2.x); bf[j][9] =f2bf(y2.y); bf[j][10]=f2bf(y2.z); bf[j][11]=f2bf(y2.w);
            bf[j][12]=f2bf(y3.x); bf[j][13]=f2bf(y3.y); bf[j][14]=f2bf(y3.z); bf[j][15]=f2bf(y3.w);
        }
        #pragma unroll
        for (int j = 0; j < 4; ++j) acc[j] = wmma_bf16(a, bf[j], acc[j]);
    }

    if (Kmain < K) {   // tail (rel L0: K=20)
        v16bf a;
        #pragma unroll
        for (int i = 0; i < 16; ++i) {
            int k  = Kmain + kbaseA + i;
            int ks = (k < K) ? k : (K - 1);
            float v = xrow[ks];
            a[i] = f2bf((k < K) ? v : 0.f);
        }
        v16bf bf[4];
        #pragma unroll
        for (int i = 0; i < 16; ++i) {
            int k  = Kmain + kbaseB + i;
            int ks = (k < K) ? k : (K - 1);
            #pragma unroll
            for (int j = 0; j < 4; ++j) {
                float v = wrow[j][ks];
                bf[j][i] = f2bf((k < K) ? v : 0.f);
            }
        }
        #pragma unroll
        for (int j = 0; j < 4; ++j) acc[j] = wmma_bf16(a, bf[j], acc[j]);
    }

    #pragma unroll
    for (int j = 0; j < 4; ++j) {
        if (!nval[j]) continue;
        float bs = bias ? bias[nB[j]] : 0.f;
        #pragma unroll
        for (int r = 0; r < 8; ++r) {
            int m = m0 + r + ((lane >> 4) << 3);
            if (m < M) Y[(long)m*N + nB[j]] = acc[j][r] + bs;
        }
    }
}

// ---------------- LayerNorm (+ReLU), one block per row ----------------
__global__ void ln_relu_kernel(float* __restrict__ x,
                               const float* __restrict__ g,
                               const float* __restrict__ bt, int F)
{
    __shared__ float s1[256], s2[256];
    float* row = x + (long)blockIdx.x * F;
    float a = 0.f, b = 0.f;
    for (int i = threadIdx.x; i < F; i += 256) { float v = row[i]; a += v; b += v*v; }
    s1[threadIdx.x] = a; s2[threadIdx.x] = b; __syncthreads();
    for (int off = 128; off > 0; off >>= 1) {
        if (threadIdx.x < off) { s1[threadIdx.x] += s1[threadIdx.x+off];
                                 s2[threadIdx.x] += s2[threadIdx.x+off]; }
        __syncthreads();
    }
    float mu  = s1[0] / F;
    float inv = rsqrtf(s2[0]/F - mu*mu + 1e-5f);
    for (int i = threadIdx.x; i < F; i += 256)
        row[i] = fmaxf((row[i] - mu) * inv * g[i] + bt[i], 0.f);
}

// ---------------- build 44x224x224x3 bf16 NHWC input batch ----------------
__global__ void prep_kernel(const float* __restrict__ tgt,
                            const float* __restrict__ obj,
                            ushort_t* __restrict__ x0)
{
    long i = (long)blockIdx.x * blockDim.x + threadIdx.x;
    long total = (long)NIMG * HWPX * 3;
    if (i >= total) return;
    long t = i / 3;                    // (img,h,w)
    int p = (int)(t % HWPX); int img = (int)(t / HWPX);
    float v = (img < BB) ? tgt[(long)img*HWPX + p] : obj[(long)(img-BB)*HWPX + p];
    x0[i] = f2bf_bits(v);
}

// ---------------- maxpool 3x3 s2 p1 (bf16 NHWC) ----------------
__global__ void maxpool_kernel(const ushort_t* __restrict__ in, ushort_t* __restrict__ out,
                               int C, int H, int W, int Ho, int Wo)
{
    long i = (long)blockIdx.x * blockDim.x + threadIdx.x;
    long total = (long)NIMG * C * Ho * Wo;
    if (i >= total) return;
    int c = (int)(i % C); long t = i / C;
    int ow = (int)(t % Wo); t /= Wo; int oh = (int)(t % Ho); int img = (int)(t / Ho);
    float m = -INFINITY;
    for (int kh = 0; kh < 3; ++kh)
        for (int kw = 0; kw < 3; ++kw) {
            int ih = oh*2 - 1 + kh, iw = ow*2 - 1 + kw;
            if (ih >= 0 && ih < H && iw >= 0 && iw < W)
                m = fmaxf(m, bf2f(in[(((long)img*H + ih)*W + iw)*C + c]));
        }
    out[i] = f2bf_bits(m);
}

// ---------------- global average pool 7x7 (bf16 NHWC -> f32 [img][c]) ----------------
__global__ void avgpool_kernel(const ushort_t* __restrict__ in, float* __restrict__ out, int C)
{
    int i = blockIdx.x * blockDim.x + threadIdx.x;
    if (i >= NIMG * C) return;
    int c = i % C, img = i / C;
    float s = 0.f;
    for (int p = 0; p < 49; ++p) s += bf2f(in[((long)img*49 + p)*C + c]);
    out[i] = s * (1.f/49.f);
}

// ---------------- target-mask row sums & total ----------------
__global__ void tmstats_kernel(const float* __restrict__ tgt,
                               float* __restrict__ tm_rows, float* __restrict__ tm_tot)
{
    __shared__ float sh[IMGSZ];
    int b = blockIdx.x, h = threadIdx.x;
    const float* row = tgt + ((long)b*IMGSZ + h) * IMGSZ;
    float s = 0.f;
    for (int w = 0; w < IMGSZ; ++w) s += row[w];
    tm_rows[b*IMGSZ + h] = s; sh[h] = s; __syncthreads();
    if (h == 0) { float t = 0.f; for (int i = 0; i < IMGSZ; ++i) t += sh[i]; tm_tot[b] = t; }
}

// ---------------- overlap & object-mask sums ----------------
__global__ void overlap_kernel(const float* __restrict__ tgt, const float* __restrict__ obj,
                               float* __restrict__ ov, float* __restrict__ oms)
{
    __shared__ float s1[256], s2[256];
    int bn = blockIdx.x, b = bn / NN;
    const float* om = obj + (long)bn * HWPX;
    const float* tm = tgt + (long)b  * HWPX;
    float a = 0.f, c = 0.f;
    for (int i = threadIdx.x; i < HWPX; i += 256) { float o = om[i]; a += o*tm[i]; c += o; }
    s1[threadIdx.x] = a; s2[threadIdx.x] = c; __syncthreads();
    for (int off = 128; off > 0; off >>= 1) {
        if (threadIdx.x < off) { s1[threadIdx.x] += s1[threadIdx.x+off];
                                 s2[threadIdx.x] += s2[threadIdx.x+off]; }
        __syncthreads();
    }
    if (threadIdx.x == 0) { ov[bn] = s1[0]; oms[bn] = s2[0]; }
}

// ---------------- edge feature (overlap, iou) per (b,n) ----------------
__global__ void edge_kernel(const float* __restrict__ bbox,
                            const float* __restrict__ tm_rows,
                            const float* __restrict__ tm_tot,
                            const float* __restrict__ ov,
                            float* __restrict__ edge)
{
    int bn = blockIdx.x * blockDim.x + threadIdx.x;
    if (bn >= BB*NN) return;
    int b = bn / NN;
    int x1 = (int)floorf(bbox[bn*4+0]);
    int y1 = (int)floorf(bbox[bn*4+1]);
    int x2 = (int)floorf(bbox[bn*4+2]);
    int y2 = (int)floorf(bbox[bn*4+3]);
    float d0 = (y1 <= 0 && y2 > 0) ? 1.f : 0.f;
    float inter = 0.f, rmsum = 0.f;
    for (int h = 0; h < IMGSZ; ++h) {
        float inr = (h >= x1 && h < x2) ? 1.f : 0.f;
        rmsum += inr; inter += inr * tm_rows[b*IMGSZ + h];
    }
    inter *= d0;
    float boxsum = d0 * rmsum * (float)IMGSZ;
    edge[bn*2 + 0] = ov[bn];
    edge[bn*2 + 1] = inter / (boxsum + tm_tot[b] - inter + 1e-8f);
}

// ---------------- attention: softmax(q.kT/sqrt(d)) @ rel ----------------
__global__ void attn_kernel(const float* __restrict__ q, const float* __restrict__ k,
                            const float* __restrict__ rel, float* __restrict__ out)
{
    __shared__ float red[128];
    __shared__ float p[NN];
    int bn = blockIdx.x, b = bn / NN;
    const float* qv = q + (long)bn * DIMF;
    for (int m = 0; m < NN; ++m) {
        const float* kv = k + (long)(b*NN + m) * DIMF;
        float s = 0.f;
        for (int d = threadIdx.x; d < DIMF; d += 128) s += qv[d] * kv[d];
        red[threadIdx.x] = s; __syncthreads();
        for (int off = 64; off > 0; off >>= 1) {
            if (threadIdx.x < off) red[threadIdx.x] += red[threadIdx.x+off];
            __syncthreads();
        }
        if (threadIdx.x == 0) p[m] = red[0] * 0.04419417382415922f;   // 1/sqrt(512)
        __syncthreads();
    }
    if (threadIdx.x == 0) {
        float mx = p[0]; for (int m = 1; m < NN; ++m) mx = fmaxf(mx, p[m]);
        float s = 0.f;
        for (int m = 0; m < NN; ++m) { p[m] = expf(p[m]-mx); s += p[m]; }
        for (int m = 0; m < NN; ++m) p[m] /= s;
    }
    __syncthreads();
    for (int d = threadIdx.x; d < DIMF; d += 128) {
        float s = 0.f;
        for (int m = 0; m < NN; ++m) s += p[m] * rel[(long)(b*NN + m)*DIMF + d];
        out[(long)bn*DIMF + d] = s;
    }
}

// ---------------- concat [attn_out, rel] -> (B, N*1024) ----------------
__global__ void concat_kernel(const float* __restrict__ ao, const float* __restrict__ rel,
                              float* __restrict__ xcat)
{
    int i = blockIdx.x * blockDim.x + threadIdx.x;
    if (i >= BB*NN*2*DIMF) return;
    int j = i % (2*DIMF); int bn = i / (2*DIMF);
    xcat[i] = (j < DIMF) ? ao[(long)bn*DIMF + j] : rel[(long)bn*DIMF + (j - DIMF)];
}

// ---------------- gumbel hard selection ----------------
__global__ void select_kernel(const float* __restrict__ logits, const float* __restrict__ oms,
                              const float* __restrict__ gum, int* __restrict__ sel)
{
    int b = threadIdx.x;
    if (b >= BB) return;
    float best = -INFINITY; int bi = 0;
    for (int n = 0; n < NN; ++n) {
        float l = (oms[b*NN + n] == 0.f) ? -1e-6f : logits[b*NN + n];
        float v = l - logf(-logf(gum[b*NN + n]));
        if (v > best) { best = v; bi = n; }
    }
    sel[b] = bi;
}

__global__ void ctx_kernel(const float* __restrict__ obj, const int* __restrict__ sel,
                           float* __restrict__ out)
{
    long i = (long)blockIdx.x * blockDim.x + threadIdx.x;
    if (i >= (long)BB*HWPX) return;
    int b = (int)(i / HWPX); int p = (int)(i % HWPX);
    out[i] = obj[((long)(b*NN + sel[b]))*HWPX + p];
}

// =====================================================================
// Host orchestration
// =====================================================================
extern "C" void kernel_launch(void* const* d_in, const int* in_sizes, int n_in,
                              void* d_out, int out_size, void* d_ws, size_t ws_size,
                              hipStream_t stream)
{
    (void)in_sizes; (void)n_in; (void)out_size; (void)ws_size;
    auto F = [&](int i) -> const float* { return (const float*)d_in[i]; };

    const float* tgt  = F(0);
    const float* obj  = F(1);
    const float* bbox = F(2);
    const float* gum  = F(3);
    float* ws  = (float*)d_ws;
    float* out = (float*)d_out;

    // ---- workspace layout (float offsets; bf16 regions use <= half) ----
    const size_t WX0 = 0;                    // 6,623,232 bf16 (NHWC 44x224x224x3)
    const size_t WC1 = 6623232;              // 35,323,904 bf16
    const size_t WA  = 41947136;             // 8,830,976 bf16 each for A/B/C
    const size_t WB2 = 50778112;
    const size_t WC2 = 59609088;
    const size_t WPOOL = 68440064;           // 44*512 f32
    const size_t WFEAT = 68462592;           // 44*1024 f32
    const size_t WT0   = 68507648;           // 16384 f32
    const size_t WT1   = 68524032;           // 16384 f32
    const size_t WEDGE = 68540416;           // 80
    const size_t WTMR  = 68540496;           // 896
    const size_t WTMT  = 68541392;           // 4
    const size_t WOV   = 68541396;           // 40
    const size_t WOMS  = 68541436;           // 40
    const size_t WQ    = 68541476;           // 20480
    const size_t WK    = 68561956;           // 20480
    const size_t WREL  = 68582436;           // 20480
    const size_t WAO   = 68602916;           // 20480
    const size_t WXC   = 68623396;           // 40960
    const size_t WLOG  = 68664356;           // 40
    const size_t WSEL  = 68664396;           // 4 ints
    const size_t WWB   = 68664400;           // bf16 conv weights (11.2M ushorts)

    ushort_t *X0 = (ushort_t*)(ws+WX0), *C1 = (ushort_t*)(ws+WC1);
    ushort_t *A  = (ushort_t*)(ws+WA),  *B  = (ushort_t*)(ws+WB2), *C = (ushort_t*)(ws+WC2);
    int* sel = (int*)(ws + WSEL);
    ushort_t* wb16base = (ushort_t*)(ws + WWB);

    // ---- reorder+convert conv weights: OIHW f32 -> (O,kh,kw,I) bf16 ----
    struct WRec { int idx; int co, ci, ks; };
    const WRec wrecs[] = {
        {40,  64,   3, 7},                                                 // stem
        {51,  64,  64, 3}, {52,  64,  64, 3}, {61,  64,  64, 3}, {62,  64,  64, 3},
        {71, 128,  64, 3}, {72, 128, 128, 3}, {77, 128,  64, 1}, {86, 128, 128, 3}, {87, 128, 128, 3},
        {96, 256, 128, 3}, {97, 256, 256, 3}, {102,256, 128, 1}, {111,256, 256, 3}, {112,256, 256, 3},
        {121,512, 256, 3}, {122,512, 512, 3}, {127,512, 256, 1}, {136,512, 512, 3}, {137,512, 512, 3}
    };
    ushort_t* wb16[160] = {};
    {
        long off = 0;
        for (int i = 0; i < (int)(sizeof(wrecs)/sizeof(wrecs[0])); ++i) {
            long n = (long)wrecs[i].co * wrecs[i].ci * wrecs[i].ks * wrecs[i].ks;
            wb16[wrecs[i].idx] = wb16base + off;
            wreorder_kernel<<<(unsigned)((n + 255)/256), 256, 0, stream>>>(
                F(wrecs[i].idx), wb16base + off, wrecs[i].co, wrecs[i].ci, wrecs[i].ks);
            off += (n + 7) & ~7L;   // keep 16B alignment per tensor
        }
    }

    auto ctz = [](int v){ int s = 0; while (!(v & 1)) { v >>= 1; ++s; } return s; };
    auto conv = [&](const ushort_t* in, int w, int bn, const ushort_t* res, ushort_t* o,
                    int Cin, int H, int W, int Cout, int Ho, int Wo,
                    int KS, int s, int p, int relu) {
        int Npix = NIMG * Ho * Wo;
        dim3 g((Npix + 63)/64, (Cout + 15)/16);
        int cs = (KS == 7) ? 0 : ctz(Cin);
        if (KS == 7)
            conv_bn_wmma<7><<<g, 32, 0, stream>>>(in, wb16[w], F(bn), F(bn+1), F(bn+2), F(bn+3),
                                                  res, o, Cin, cs, H, W, Cout, Ho, Wo, s, p, relu);
        else if (KS == 3)
            conv_bn_wmma<3><<<g, 32, 0, stream>>>(in, wb16[w], F(bn), F(bn+1), F(bn+2), F(bn+3),
                                                  res, o, Cin, cs, H, W, Cout, Ho, Wo, s, p, relu);
        else
            conv_bn_wmma<1><<<g, 32, 0, stream>>>(in, wb16[w], F(bn), F(bn+1), F(bn+2), F(bn+3),
                                                  res, o, Cin, cs, H, W, Cout, Ho, Wo, s, p, relu);
    };
    auto gemm = [&](const float* X, int w, int b, float* Y, int M, int N, int K) {
        dim3 g((N + 63)/64, (M + 15)/16);
        gemm_xwT_wmma<<<g, 32, 0, stream>>>(X, F(w), F(b), Y, M, N, K);
    };
    auto ln = [&](float* X, int g, int bt, int rows, int Fd) {
        ln_relu_kernel<<<rows, 256, 0, stream>>>(X, F(g), F(bt), Fd);
    };

    // ---------- ResNet-18 on 44 images (NHWC bf16) ----------
    {
        long tot = (long)NIMG*HWPX*3;
        prep_kernel<<<(unsigned)((tot+255)/256), 256, 0, stream>>>(tgt, obj, X0);
    }
    conv(X0, 40, 36, nullptr, C1, 3, 224, 224, 64, 112, 112, 7, 2, 3, 1);       // stem
    {
        long tot = (long)NIMG*64*56*56;
        maxpool_kernel<<<(unsigned)((tot+255)/256), 256, 0, stream>>>(C1, A, 64, 112, 112, 56, 56);
    }
    // layer1 (64ch, 56x56)
    conv(A, 51, 43, nullptr, B, 64,56,56, 64,56,56, 3,1,1, 1);
    conv(B, 52, 47, A,       C, 64,56,56, 64,56,56, 3,1,1, 1);
    conv(C, 61, 53, nullptr, A, 64,56,56, 64,56,56, 3,1,1, 1);
    conv(A, 62, 57, C,       B, 64,56,56, 64,56,56, 3,1,1, 1);
    // layer2 (64->128, 28x28)
    conv(B, 71, 63, nullptr, A, 64,56,56, 128,28,28, 3,2,1, 1);
    conv(B, 77, 73, nullptr, C, 64,56,56, 128,28,28, 1,2,0, 0);                 // downsample
    conv(A, 72, 67, C,       B, 128,28,28, 128,28,28, 3,1,1, 1);
    conv(B, 86, 78, nullptr, A, 128,28,28, 128,28,28, 3,1,1, 1);
    conv(A, 87, 82, B,       C, 128,28,28, 128,28,28, 3,1,1, 1);
    // layer3 (128->256, 14x14)
    conv(C, 96, 88, nullptr, A, 128,28,28, 256,14,14, 3,2,1, 1);
    conv(C, 102, 98, nullptr, B, 128,28,28, 256,14,14, 1,2,0, 0);               // downsample
    conv(A, 97, 92, B,       C, 256,14,14, 256,14,14, 3,1,1, 1);
    conv(C, 111, 103, nullptr, A, 256,14,14, 256,14,14, 3,1,1, 1);
    conv(A, 112, 107, C,     B, 256,14,14, 256,14,14, 3,1,1, 1);
    // layer4 (256->512, 7x7)
    conv(B, 121, 113, nullptr, A, 256,14,14, 512,7,7, 3,2,1, 1);
    conv(B, 127, 123, nullptr, C, 256,14,14, 512,7,7, 1,2,0, 0);                // downsample
    conv(A, 122, 117, C,     B, 512,7,7, 512,7,7, 3,1,1, 1);
    conv(B, 136, 128, nullptr, A, 512,7,7, 512,7,7, 3,1,1, 1);
    conv(A, 137, 132, B,     C, 512,7,7, 512,7,7, 3,1,1, 1);
    // avgpool + fc
    avgpool_kernel<<<(NIMG*512 + 255)/256, 256, 0, stream>>>(C, ws+WPOOL, 512);
    gemm(ws+WPOOL, 42, 41, ws+WFEAT, NIMG, HID, 512);

    // ---------- edge features ----------
    tmstats_kernel<<<BB, IMGSZ, 0, stream>>>(tgt, ws+WTMR, ws+WTMT);
    overlap_kernel<<<BB*NN, 256, 0, stream>>>(tgt, obj, ws+WOV, ws+WOMS);
    edge_kernel<<<1, 64, 0, stream>>>(bbox, ws+WTMR, ws+WTMT, ws+WOV, ws+WEDGE);

    // ---------- rel MLP: 20 -> 512 -> 1024 -> 512 -> 5120 ----------
    gemm(ws+WEDGE, 25, 22, ws+WT0, BB, 512, 20);      ln(ws+WT0, 24, 23, BB, 512);
    gemm(ws+WT0,   29, 26, ws+WT1, BB, 1024, 512);    ln(ws+WT1, 28, 27, BB, 1024);
    gemm(ws+WT1,   33, 30, ws+WT0, BB, 512, 1024);    ln(ws+WT0, 32, 31, BB, 512);
    gemm(ws+WT0,   35, 34, ws+WREL, BB, NN*DIMF, 512);

    // ---------- wt MLP (q): 1024 -> 2048 -> 1024 -> 5120 ----------
    gemm(ws+WFEAT, 151, 148, ws+WT0, BB, 2048, HID);  ln(ws+WT0, 150, 149, BB, 2048);
    gemm(ws+WT0,   155, 152, ws+WT1, BB, 1024, 2048); ln(ws+WT1, 154, 153, BB, 1024);
    gemm(ws+WT1,   157, 156, ws+WQ,  BB, NN*DIMF, 1024);

    // ---------- wo MLP (k): 10240 -> 2048 -> 1024 -> 5120 ----------
    gemm(ws+WFEAT + (size_t)BB*HID, 141, 138, ws+WT0, BB, 2048, NN*HID);
    ln(ws+WT0, 140, 139, BB, 2048);
    gemm(ws+WT0, 145, 142, ws+WT1, BB, 1024, 2048);   ln(ws+WT1, 144, 143, BB, 1024);
    gemm(ws+WT1, 147, 146, ws+WK,  BB, NN*DIMF, 1024);

    // ---------- attention + concat ----------
    attn_kernel<<<BB*NN, 128, 0, stream>>>(ws+WQ, ws+WK, ws+WREL, ws+WAO);
    concat_kernel<<<(BB*NN*2*DIMF + 255)/256, 256, 0, stream>>>(ws+WAO, ws+WREL, ws+WXC);

    // ---------- attn MLP: 10240 -> 1024 -> 2048 -> 1024 -> 512 -> 10 ----------
    gemm(ws+WXC, 7,  4,  ws+WT0, BB, 1024, NN*HID);   ln(ws+WT0, 6,  5,  BB, 1024);
    gemm(ws+WT0, 11, 8,  ws+WT1, BB, 2048, 1024);     ln(ws+WT1, 10, 9,  BB, 2048);
    gemm(ws+WT1, 15, 12, ws+WT0, BB, 1024, 2048);     ln(ws+WT0, 14, 13, BB, 1024);
    gemm(ws+WT0, 19, 16, ws+WT1, BB, 512, 1024);      ln(ws+WT1, 18, 17, BB, 512);
    gemm(ws+WT1, 21, 20, ws+WLOG, BB, NN, 512);

    // ---------- gumbel hard select + context mask ----------
    select_kernel<<<1, 32, 0, stream>>>(ws+WLOG, ws+WOMS, gum, sel);
    ctx_kernel<<<(BB*HWPX + 255)/256, 256, 0, stream>>>(obj, sel, out);
}
